// TemporalGNN_18683107737890
// MI455X (gfx1250) — matrix operand order repaired and checked by reading, not compiled, not run on hardware
//
#include <hip/hip_runtime.h>
#include <hip/hip_bf16.h>
#include <math.h>

// ---------------------------------------------------------------------------
// TemporalGNN on gfx1250: bf16 WMMA for the two feature GEMMs (B pre-packed
// into WMMA fragment order so the inner loop is b128 loads + v_wmma only),
// bandwidth-bound VALU kernels + float atomics for edge softmax/scatter.
// ---------------------------------------------------------------------------

#define N_NODES_C 100000
#define N_EDGES_C 1600000
#define G_C       256
#define NEG_SLOPE 0.2f

typedef __attribute__((ext_vector_type(16))) __bf16 v16bf;
typedef __attribute__((ext_vector_type(8)))  float  v8f;

// ---------------- small utility kernels ----------------

__global__ void k_fill(float* p, float v, long long n) {
  long long i = (long long)blockIdx.x * blockDim.x + threadIdx.x;
  if (i < n) p[i] = v;
}

__global__ void k_f32_to_bf16(const float* __restrict__ in, __bf16* __restrict__ out, long long n) {
  long long i = (long long)blockIdx.x * blockDim.x + threadIdx.x;
  if (i < n) out[i] = (__bf16)in[i];
}

__global__ void k_edge_w(const float* __restrict__ ea, const float* __restrict__ Wt,
                         const float* __restrict__ bt, float* __restrict__ ew, long long E) {
  long long e = (long long)blockIdx.x * blockDim.x + threadIdx.x;
  if (e >= E) return;
  float z = ea[2 * e] * Wt[0] + ea[2 * e + 1] * Wt[1] + bt[0];
  ew[e] = 1.0f / (1.0f + expf(-z));
}

// ---------------- B pre-pack: f32 [128 x N] -> bf16 WMMA fragment order ------
// Fragment order: Bp[((nt*4 + kk)*32 + lane)*16 + i] =
//                 B[(kk*32 + (i>>3)*16 + (lane>>4)*8 + (i&7)) * N + nt*16 + (lane&15)]
// so each lane's 16 elements of a K-step fragment are contiguous (32B aligned).
__global__ void k_pack_B(const float* __restrict__ B, __bf16* __restrict__ Bp, int N) {
  int idx = blockIdx.x * blockDim.x + threadIdx.x;   // total = N*128
  if (idx >= N * 128) return;
  int i    = idx & 15;
  int lane = (idx >> 4) & 31;
  int kk   = (idx >> 9) & 3;
  int nt   = idx >> 11;
  int krow = kk * 32 + (i >> 3) * 16 + (lane >> 4) * 8 + (i & 7);
  Bp[idx] = (__bf16)B[krow * N + nt * 16 + (lane & 15)];
}

// ---------------- WMMA GEMM: C[M x N] = A[M x 128] * B[128 x N] ----------------
// One block per 16-row tile; wave w owns columns [16w, 16w+16). K = 128 fixed.
// Bp is in packed fragment order (see k_pack_B).
template <int NTILES>
__global__ __launch_bounds__(32 * NTILES)
void k_gemm_bf16(const __bf16* __restrict__ A, const __bf16* __restrict__ Bp,
                 float* __restrict__ C, int N) {
  __shared__ __bf16 Asm[16 * 128];
  const int tid  = threadIdx.x;
  const int lane = tid & 31;
  const int wave = tid >> 5;
  const int m0   = blockIdx.x * 16;

  // cooperative, coalesced A tile load (2048 bf16 = 1024 dwords)
  const unsigned* Ag   = (const unsigned*)(A + (size_t)m0 * 128);
  unsigned*       As32 = (unsigned*)Asm;
  for (int i = tid; i < 1024; i += 32 * NTILES) As32[i] = Ag[i];
  __syncthreads();

  const int mrow = lane & 15;
  const int kh   = lane >> 4;                 // half-wave K split
  const int ncol = wave * 16 + (lane & 15);
  const v16bf* Bpv = (const v16bf*)Bp;        // 32B-aligned fragments

  v8f acc = {};
#pragma unroll
  for (int kk = 0; kk < 4; ++kk) {            // K in steps of 32
    const int kb = kk * 32;
    v16bf a;
#pragma unroll
    for (int i = 0; i < 8; ++i) {             // ISA 16-bit A 16x32 layout
      a[i]     = Asm[mrow * 128 + kb + kh * 8 + i];        // ds_load_b128
      a[8 + i] = Asm[mrow * 128 + kb + 16 + kh * 8 + i];   // ds_load_b128
    }
    v16bf b = Bpv[(wave * 4 + kk) * 32 + lane];            // 2x global_load_b128
    acc = __builtin_amdgcn_wmma_f32_16x16x32_bf16(false, a, false, b, (short)0, acc,
                                                  false, false);
  }
#pragma unroll
  for (int r = 0; r < 8; ++r) {               // D layout: VGPR r -> row r + 8*kh
    const int m = m0 + r + kh * 8;
    C[(size_t)m * N + ncol] = acc[r];
  }
}

// ---------------- attention score dot products ----------------
// h layout [N, H*64]; idx = n*H + head -> row offset idx*64
__global__ void k_att(const float* __restrict__ h, const float* __restrict__ att_s,
                      const float* __restrict__ att_d, float* __restrict__ as_,
                      float* __restrict__ ad_, long long NH, int H) {
  long long idx = (long long)blockIdx.x * blockDim.x + threadIdx.x;
  if (idx >= NH) return;
  const int hh = (int)(idx % H);
  const float* hp = h + (size_t)idx * 64;
  float s = 0.f, d = 0.f;
#pragma unroll
  for (int c = 0; c < 64; ++c) {
    float v = hp[c];
    s += v * att_s[hh * 64 + c];
    d += v * att_d[hh * 64 + c];
  }
  as_[idx] = s;
  ad_[idx] = d;
}

__device__ inline void atomicMaxF(float* addr, float val) {
  if (val >= 0.f) atomicMax((int*)addr, __float_as_int(val));
  else            atomicMin((unsigned int*)addr, __float_as_uint(val));
}

__global__ void k_logit(const int* __restrict__ src, const int* __restrict__ dst,
                        const float* __restrict__ as_, const float* __restrict__ ad_,
                        float* __restrict__ logit, float* __restrict__ maxb,
                        long long EH, int H) {
  long long idx = (long long)blockIdx.x * blockDim.x + threadIdx.x;
  if (idx >= EH) return;
  long long e = idx / H;
  int hh = (int)(idx % H);
  int s = src[e], d = dst[e];
  float l = as_[(size_t)s * H + hh] + ad_[(size_t)d * H + hh];
  l = l > 0.f ? l : NEG_SLOPE * l;
  logit[idx] = l;
  atomicMaxF(&maxb[(size_t)d * H + hh], l);
}

__global__ void k_expsum(const int* __restrict__ dst, float* __restrict__ logit,
                         const float* __restrict__ maxb, float* __restrict__ denom,
                         long long EH, int H) {
  long long idx = (long long)blockIdx.x * blockDim.x + threadIdx.x;
  if (idx >= EH) return;
  long long e = idx / H;
  int hh = (int)(idx % H);
  int d = dst[e];
  float ex = expf(logit[idx] - maxb[(size_t)d * H + hh]);
  logit[idx] = ex;
  atomicAdd(&denom[(size_t)d * H + hh], ex);
}

__global__ void k_coef(const int* __restrict__ dst, const float* __restrict__ ex,
                       const float* __restrict__ denom, const float* __restrict__ ew,
                       float* __restrict__ coef, long long EH, int H) {
  long long idx = (long long)blockIdx.x * blockDim.x + threadIdx.x;
  if (idx >= EH) return;
  long long e = idx / H;
  int hh = (int)(idx % H);
  int d = dst[e];
  coef[idx] = ex[idx] / (denom[(size_t)d * H + hh] + 1e-16f) * ew[e];
}

__global__ void k_scatter(const int* __restrict__ src, const int* __restrict__ dst,
                          const float* __restrict__ hmat, const float* __restrict__ coef,
                          float* __restrict__ out, long long total, int HC, int H) {
  long long idx = (long long)blockIdx.x * blockDim.x + threadIdx.x;
  if (idx >= total) return;
  long long e = idx / HC;
  int j = (int)(idx - e * HC);
  int hh = j >> 6;                               // C == 64 for both layers
  float v = hmat[(size_t)src[e] * HC + j] * coef[e * H + hh];
  atomicAdd(&out[(size_t)dst[e] * HC + j], v);
}

__global__ void k_bias_relu_bf16(const float* __restrict__ in, const float* __restrict__ bias,
                                 __bf16* __restrict__ out, long long n, int HC) {
  long long i = (long long)blockIdx.x * blockDim.x + threadIdx.x;
  if (i >= n) return;
  float v = in[i] + bias[i % HC];
  out[i] = (__bf16)(v > 0.f ? v : 0.f);
}

__global__ void k_bias_relu_f32(const float* __restrict__ in, const float* __restrict__ bias,
                                float* __restrict__ out, long long n, int HC) {
  long long i = (long long)blockIdx.x * blockDim.x + threadIdx.x;
  if (i >= n) return;
  float v = in[i] + bias[i % HC];
  out[i] = v > 0.f ? v : 0.f;
}

__global__ void k_pool(const float* __restrict__ h, const int* __restrict__ batch,
                       float* __restrict__ gsum, float* __restrict__ gcnt, long long n) {
  long long i = (long long)blockIdx.x * blockDim.x + threadIdx.x;
  if (i >= n) return;
  int nn = (int)(i >> 6);
  int c  = (int)(i & 63);
  int g  = batch[nn];
  atomicAdd(&gsum[(size_t)g * 64 + c], h[i]);
  if (c == 0) atomicAdd(&gcnt[g], 1.0f);
}

__global__ void k_head(const float* __restrict__ gsum, const float* __restrict__ gcnt,
                       const float* __restrict__ Wl1, const float* __restrict__ bl1,
                       const float* __restrict__ Wl2, const float* __restrict__ bl2,
                       float* __restrict__ out) {
  int g = threadIdx.x;
  if (g >= G_C) return;
  float cnt = gcnt[g];
  cnt = cnt > 1.f ? cnt : 1.f;
  float gv[64];
#pragma unroll
  for (int c = 0; c < 64; ++c) gv[c] = gsum[(size_t)g * 64 + c] / cnt;
  float o0 = bl2[0], o1 = bl2[1];
  for (int j = 0; j < 64; ++j) {
    float t = bl1[j];
#pragma unroll
    for (int c = 0; c < 64; ++c) t += gv[c] * Wl1[c * 64 + j];
    t = t > 0.f ? t : 0.f;
    o0 += t * Wl2[j * 2 + 0];
    o1 += t * Wl2[j * 2 + 1];
  }
  out[g * 2 + 0] = o0;
  out[g * 2 + 1] = o1;
}

// ---------------- host orchestration ----------------

static inline unsigned nblk(long long n, int bs) { return (unsigned)((n + bs - 1) / bs); }

extern "C" void kernel_launch(void* const* d_in, const int* in_sizes, int n_in,
                              void* d_out, int out_size, void* d_ws, size_t ws_size,
                              hipStream_t stream) {
  const float* x      = (const float*)d_in[0];
  const int*   eidx   = (const int*)d_in[1];
  const float* eattr  = (const float*)d_in[2];
  const int*   batch  = (const int*)d_in[3];
  const float* Wt     = (const float*)d_in[4];
  const float* bt     = (const float*)d_in[5];
  const float* W1     = (const float*)d_in[6];
  const float* atts1  = (const float*)d_in[7];
  const float* attd1  = (const float*)d_in[8];
  const float* b1     = (const float*)d_in[9];
  const float* W2     = (const float*)d_in[10];
  const float* atts2  = (const float*)d_in[11];
  const float* attd2  = (const float*)d_in[12];
  const float* b2     = (const float*)d_in[13];
  const float* Wl1    = (const float*)d_in[14];
  const float* bl1    = (const float*)d_in[15];
  const float* Wl2    = (const float*)d_in[16];
  const float* bl2    = (const float*)d_in[17];
  float* out = (float*)d_out;

  const long long N = N_NODES_C, E = N_EDGES_C;
  const int* src = eidx;
  const int* dst = eidx + E;

  // workspace carve-up (regions reused across phases)
  char* base = (char*)d_ws;
  size_t off = 0;
  auto carve = [&](size_t bytes) { void* p = base + off; off = (off + bytes + 255) & ~(size_t)255; return p; };
  float*  regA  = (float*)carve((size_t)N * 128 * 4);   // h1, later out2
  float*  regB  = (float*)carve((size_t)N * 128 * 4);   // out1, later h2 + h2r
  __bf16* regC  = (__bf16*)carve((size_t)N * 128 * 2);  // x_bf16, later h1r_bf16
  __bf16* w1p   = (__bf16*)carve(128 * 128 * 2);        // W1 packed fragments
  __bf16* w2p   = (__bf16*)carve(128 * 64 * 2);         // W2 packed fragments
  float*  as1   = (float*)carve((size_t)N * 2 * 4);
  float*  ad1   = (float*)carve((size_t)N * 2 * 4);
  float*  as2   = (float*)carve((size_t)N * 4);
  float*  ad2   = (float*)carve((size_t)N * 4);
  float*  ew    = (float*)carve((size_t)E * 4);
  float*  maxb  = (float*)carve((size_t)N * 2 * 4);
  float*  denom = (float*)carve((size_t)N * 2 * 4);
  float*  elog  = (float*)carve((size_t)E * 2 * 4);
  float*  coef  = (float*)carve((size_t)E * 2 * 4);
  float*  gsum  = (float*)carve((size_t)G_C * 64 * 4);
  float*  gcnt  = (float*)carve((size_t)G_C * 4);

  float*  h1   = regA;                       // [N,128]  x@W1
  float*  out1 = regB;                       // [N,128]  layer1 aggregate
  __bf16* xb   = regC;                       // [N,128]  bf16 x
  __bf16* h1rb = regC;                       // reuses xb after gemm1
  float*  h2   = regB;                       // [N,64]   h1r@W2 (out1 dead)
  float*  h2r  = regB + (size_t)N * 64;      // [N,64]   relu(out2+b2)
  float*  out2 = regA;                       // [N,64]   layer2 aggregate (h1 dead)

  const int BS = 256;

  // precision conversion, weight packing, edge weights
  k_f32_to_bf16<<<nblk(N * 128, BS), BS, 0, stream>>>(x, xb, N * 128);
  k_pack_B<<<nblk(128 * 128, BS), BS, 0, stream>>>(W1, w1p, 128);
  k_pack_B<<<nblk(64 * 128, BS), BS, 0, stream>>>(W2, w2p, 64);
  k_edge_w<<<nblk(E, BS), BS, 0, stream>>>(eattr, Wt, bt, ew, E);

  // ---- layer 1: GATConv(128 -> 64, heads=2) ----
  k_gemm_bf16<8><<<dim3(N_NODES_C / 16), 256, 0, stream>>>(xb, w1p, h1, 128);
  k_att<<<nblk(N * 2, BS), BS, 0, stream>>>(h1, atts1, attd1, as1, ad1, N * 2, 2);
  k_fill<<<nblk(N * 2, BS), BS, 0, stream>>>(maxb, -INFINITY, N * 2);
  k_fill<<<nblk(N * 2, BS), BS, 0, stream>>>(denom, 0.f, N * 2);
  k_fill<<<nblk(N * 128, BS), BS, 0, stream>>>(out1, 0.f, N * 128);
  k_logit<<<nblk(E * 2, BS), BS, 0, stream>>>(src, dst, as1, ad1, elog, maxb, E * 2, 2);
  k_expsum<<<nblk(E * 2, BS), BS, 0, stream>>>(dst, elog, maxb, denom, E * 2, 2);
  k_coef<<<nblk(E * 2, BS), BS, 0, stream>>>(dst, elog, denom, ew, coef, E * 2, 2);
  k_scatter<<<nblk(E * 128, BS), BS, 0, stream>>>(src, dst, h1, coef, out1, E * 128, 128, 2);
  k_bias_relu_bf16<<<nblk(N * 128, BS), BS, 0, stream>>>(out1, b1, h1rb, N * 128, 128);

  // ---- layer 2: GATConv(128 -> 64, heads=1) ----
  k_gemm_bf16<4><<<dim3(N_NODES_C / 16), 128, 0, stream>>>(h1rb, w2p, h2, 64);
  k_att<<<nblk(N, BS), BS, 0, stream>>>(h2, atts2, attd2, as2, ad2, N, 1);
  k_fill<<<nblk(N, BS), BS, 0, stream>>>(maxb, -INFINITY, N);
  k_fill<<<nblk(N, BS), BS, 0, stream>>>(denom, 0.f, N);
  k_fill<<<nblk(N * 64, BS), BS, 0, stream>>>(out2, 0.f, N * 64);
  k_logit<<<nblk(E, BS), BS, 0, stream>>>(src, dst, as2, ad2, elog, maxb, E, 1);
  k_expsum<<<nblk(E, BS), BS, 0, stream>>>(dst, elog, maxb, denom, E, 1);
  k_coef<<<nblk(E, BS), BS, 0, stream>>>(dst, elog, denom, ew, coef, E, 1);
  k_scatter<<<nblk(E * 64, BS), BS, 0, stream>>>(src, dst, h2, coef, out2, E * 64, 64, 1);
  k_bias_relu_f32<<<nblk(N * 64, BS), BS, 0, stream>>>(out2, b2, h2r, N * 64, 64);

  // ---- pooling + MLP head ----
  k_fill<<<nblk(G_C * 64, BS), BS, 0, stream>>>(gsum, 0.f, G_C * 64);
  k_fill<<<nblk(G_C, BS), BS, 0, stream>>>(gcnt, 0.f, G_C);
  k_pool<<<nblk(N * 64, BS), BS, 0, stream>>>(h2r, batch, gsum, gcnt, N * 64);
  k_head<<<1, 256, 0, stream>>>(gsum, gcnt, Wl1, bl1, Wl2, bl2, out);
}
